// G_Att_Net_17669495455829
// MI455X (gfx1250) — matrix operand order repaired and checked by reading, not compiled, not run on hardware
//
#include <hip/hip_runtime.h>
#include <hip/hip_bf16.h>
#include <stdint.h>

// Problem dims (fixed by reference)
#define N_ROWS 50000
#define D_IN   1024
#define D_C    512
#define D_H    256

// GEMM tiling: 128x128 block tile, K-step 32 (one wmma_f32_16x16x32_bf16),
// 256 threads = 8 wave32s, each wave computes a 64(M) x 32(N) sub-tile.
// Grid: blockIdx.x = N tile (fast-varying, so same-row blocks share L2-hot A rows),
//       blockIdx.y = M tile.
#define BM 128
#define BN 128
#define BK 32
#define THREADS 256
#define LDSS 40   // BK + 8 ushort pad to spread LDS banks

typedef __attribute__((ext_vector_type(16))) __bf16 v16bf;
typedef __attribute__((ext_vector_type(8)))  float  v8f;
typedef int v4i_gv __attribute__((vector_size(16)));   // matches builtin's V4i

// ---------------- gfx1250 async global->LDS copy path ----------------
#if defined(__gfx1250__) && __has_builtin(__builtin_amdgcn_global_load_async_to_lds_b128)
#define HAVE_ASYNC_LDS 1
#else
#define HAVE_ASYNC_LDS 0
#endif

__device__ __forceinline__ void cp_b128_async(void* lptr, const void* gptr) {
#if HAVE_ASYNC_LDS
  __builtin_amdgcn_global_load_async_to_lds_b128(
      (__attribute__((address_space(1))) v4i_gv*)gptr,
      (__attribute__((address_space(3))) v4i_gv*)lptr, 0, 0);
#else
  *(uint4*)lptr = *(const uint4*)gptr;
#endif
}

__device__ __forceinline__ void wait_async_lds() {
#if HAVE_ASYNC_LDS
#if __has_builtin(__builtin_amdgcn_s_wait_asynccnt)
  __builtin_amdgcn_s_wait_asynccnt(0);
#else
  asm volatile("s_wait_asynccnt 0x0" ::: "memory");
#endif
#endif
}

// ---------------- scalar helpers ----------------
__device__ __forceinline__ unsigned short f2bf(float x) {
  union { float f; unsigned int u; } c; c.f = x;
  unsigned int u = c.u;
  return (unsigned short)((u + 0x7FFFu + ((u >> 16) & 1u)) >> 16); // RNE
}
__device__ __forceinline__ float bf2f(unsigned short h) {
  union { float f; unsigned int u; } c; c.u = ((unsigned int)h) << 16;
  return c.f;
}

// packed f32x2 -> bf16x2 (native packed cvt if the toolchain exposes it)
__device__ __forceinline__ unsigned int pack_bf16x2(float a, float b) {
#if defined(__gfx1250__) && __has_builtin(__builtin_amdgcn_cvt_pk_bf16_f32)
  typedef __attribute__((ext_vector_type(2))) __bf16 v2bf;
  v2bf r = __builtin_amdgcn_cvt_pk_bf16_f32(a, b);
  return __builtin_bit_cast(unsigned int, r);
#else
  return (unsigned int)f2bf(a) | ((unsigned int)f2bf(b) << 16);
#endif
}

struct Frag32B { uint4 lo; uint4 hi; };
__device__ __forceinline__ v16bf ld_frag(const unsigned short* p0, const unsigned short* p1) {
  Frag32B s;
  s.lo = *(const uint4*)p0;   // ds_load_b128
  s.hi = *(const uint4*)p1;   // ds_load_b128
  return __builtin_bit_cast(v16bf, s);
}

// ---------------- epilogue helper ----------------
// C/D layout: VGPR r holds M = r + 8*(lane>>4), N = lane&15.
// ACT: 0 relu (f32 out + bf16 ws), 1 none->bf16, 2 tanh->bf16, 3 sigmoid*gate->bf16
template <int ACT>
__device__ __forceinline__ void epilogue(v8f acc[4][2], int blk_m, int blk_n, int wm, int wn,
                                         int l15, int lhi,
                                         const float* __restrict__ bias,
                                         float* __restrict__ out_f32,
                                         unsigned short* __restrict__ out_bf16,
                                         const unsigned short* __restrict__ gate,
                                         int M, int Nn) {
#pragma unroll
  for (int i = 0; i < 4; ++i) {
    int mbase = blk_m + wm + i * 16 + 8 * lhi;
#pragma unroll
    for (int j = 0; j < 2; ++j) {
      int n = blk_n + wn + j * 16 + l15;
      float bv = bias[n];
#pragma unroll
      for (int r = 0; r < 8; ++r) {
        int m = mbase + r;
        if (m < M) {
          float x = acc[i][j][r] + bv;
          size_t idx = (size_t)m * Nn + n;
          if (ACT == 0) {
            x = x > 0.f ? x : 0.f;
            out_f32[idx]  = x;
            out_bf16[idx] = f2bf(x);
          } else if (ACT == 1) {
            out_bf16[idx] = f2bf(x);
          } else if (ACT == 2) {
            out_bf16[idx] = f2bf(tanhf(x));
          } else {
            float s = 1.f / (1.f + __expf(-x));
            out_bf16[idx] = f2bf(s * bf2f(gate[idx]));
          }
        }
      }
    }
  }
}

#define COMPUTE_STEP(cur)                                                              \
  {                                                                                    \
    v16bf bfr[2];                                                                      \
    _Pragma("unroll")                                                                  \
    for (int j = 0; j < 2; ++j) {                                                      \
      const unsigned short* bp = &Bs[cur][(wn + j * 16 + l15) * LDSS + lhi * 8];       \
      bfr[j] = ld_frag(bp, bp + 16);                                                   \
    }                                                                                  \
    _Pragma("unroll")                                                                  \
    for (int i = 0; i < 4; ++i) {                                                      \
      const unsigned short* ap = &As[cur][(wm + i * 16 + l15) * LDSS + lhi * 8];       \
      v16bf afr = ld_frag(ap, ap + 16);                                                \
      _Pragma("unroll")                                                                \
      for (int j = 0; j < 2; ++j) {                                                    \
        acc[i][j] = __builtin_amdgcn_wmma_f32_16x16x32_bf16(                           \
            false, afr, false, bfr[j], (short)0, acc[i][j], false, false);             \
      }                                                                                \
    }                                                                                  \
  }

// ---------------- Layer 1: h = relu(X @ Wc + bc), X f32 ----------------
// A: global f32 -> regs -> cvt bf16 -> LDS (double-buffered, reg prefetch)
// B: K-major bf16 weights Bt[n][k] -> LDS via async b128 copies
__global__ __launch_bounds__(THREADS)
void gemm_l1_kernel(const float* __restrict__ X,
                    const unsigned short* __restrict__ Bt,   // [Nn][K] bf16 (K-major)
                    const float* __restrict__ bias,
                    float* __restrict__ out_f32,
                    unsigned short* __restrict__ out_bf16,
                    int M, int Nn, int K)
{
  __shared__ unsigned short As[2][BM * LDSS];
  __shared__ unsigned short Bs[2][BN * LDSS];

  const int t = threadIdx.x;
  const int lane = t & 31, wid = t >> 5;
  const int wm = (wid & 1) * 64, wn = (wid >> 1) * 32;
  const int blk_n = blockIdx.x * BN, blk_m = blockIdx.y * BM;  // N tile fast-varying
  const int l15 = lane & 15, lhi = lane >> 4;

  v8f acc[4][2];
#pragma unroll
  for (int i = 0; i < 4; ++i)
#pragma unroll
    for (int j = 0; j < 2; ++j)
      acc[i][j] = v8f{0.f, 0.f, 0.f, 0.f, 0.f, 0.f, 0.f, 0.f};

  // copy roles: thread t handles 16 K-elements (chalf*16..+15) of one row
  const int crow  = t >> 1;     // 0..127
  const int chalf = t & 1;      // 0/1
  int arow = blk_m + crow; if (arow >= M) arow = M - 1;
  const float*          ag = X  + (size_t)arow * K + chalf * 16;
  const unsigned short* bg = Bt + (size_t)(blk_n + crow) * K + chalf * 16;
  const int niter = K / BK;

  float4 ar[4];
  { // prologue: A regs for kt=0, async B tile 0
    const float4* s = (const float4*)ag;
#pragma unroll
    for (int c = 0; c < 4; ++c) ar[c] = s[c];
    unsigned short* bd = &Bs[0][crow * LDSS + chalf * 16];
    cp_b128_async(bd, bg);
    cp_b128_async(bd + 8, bg + 8);
  }

  for (int it = 0; it < niter; ++it) {
    const int cur = it & 1;
    wait_async_lds();
    __syncthreads();   // Bs[cur] published; everyone finished compute(it-1)
    { // store current A regs (cvt f32->bf16) into As[cur]
      unsigned int tmp[8];
#pragma unroll
      for (int c = 0; c < 4; ++c) {
        tmp[c * 2 + 0] = pack_bf16x2(ar[c].x, ar[c].y);
        tmp[c * 2 + 1] = pack_bf16x2(ar[c].z, ar[c].w);
      }
      uint4* d = (uint4*)&As[cur][crow * LDSS + chalf * 16];
      d[0] = ((const uint4*)tmp)[0];
      d[1] = ((const uint4*)tmp)[1];
    }
    if (it + 1 < niter) {  // prefetch next A regs + async next B tile
      const float4* s = (const float4*)(ag + (it + 1) * BK);
#pragma unroll
      for (int c = 0; c < 4; ++c) ar[c] = s[c];
      const unsigned short* bsrc = bg + (it + 1) * BK;
      unsigned short* bd = &Bs[1 - cur][crow * LDSS + chalf * 16];
      cp_b128_async(bd, bsrc);
      cp_b128_async(bd + 8, bsrc + 8);
    }
    __syncthreads();   // As[cur] visible
    COMPUTE_STEP(cur)
  }
  epilogue<0>(acc, blk_m, blk_n, wm, wn, l15, lhi, bias, out_f32, out_bf16, nullptr, M, Nn);
}

// ---------------- Layers 2-5: bf16 A, fully async double-buffered ----------------
template <int ACT>
__global__ __launch_bounds__(THREADS)
void gemm_bf16_kernel(const unsigned short* __restrict__ Ab,  // [M][K] bf16
                      const unsigned short* __restrict__ Bt,  // [Nn][K] bf16 (K-major)
                      const float* __restrict__ bias,
                      unsigned short* __restrict__ out_bf16,
                      const unsigned short* __restrict__ gate,
                      int M, int Nn, int K)
{
  __shared__ unsigned short As[2][BM * LDSS];
  __shared__ unsigned short Bs[2][BN * LDSS];

  const int t = threadIdx.x;
  const int lane = t & 31, wid = t >> 5;
  const int wm = (wid & 1) * 64, wn = (wid >> 1) * 32;
  const int blk_n = blockIdx.x * BN, blk_m = blockIdx.y * BM;  // N tile fast-varying
  const int l15 = lane & 15, lhi = lane >> 4;

  v8f acc[4][2];
#pragma unroll
  for (int i = 0; i < 4; ++i)
#pragma unroll
    for (int j = 0; j < 2; ++j)
      acc[i][j] = v8f{0.f, 0.f, 0.f, 0.f, 0.f, 0.f, 0.f, 0.f};

  const int crow  = t >> 1;
  const int chalf = t & 1;
  int arow = blk_m + crow; if (arow >= M) arow = M - 1;
  const unsigned short* ag = Ab + (size_t)arow * K + chalf * 16;
  const unsigned short* bg = Bt + (size_t)(blk_n + crow) * K + chalf * 16;
  const int niter = K / BK;

  auto issue = [&](int buf, int kt) {
    unsigned short* ad = &As[buf][crow * LDSS + chalf * 16];
    unsigned short* bd = &Bs[buf][crow * LDSS + chalf * 16];
    cp_b128_async(ad,     ag + kt);
    cp_b128_async(ad + 8, ag + kt + 8);
    cp_b128_async(bd,     bg + kt);
    cp_b128_async(bd + 8, bg + kt + 8);
  };

  issue(0, 0);
  for (int it = 0; it < niter; ++it) {
    const int cur = it & 1;
    wait_async_lds();       // own asyncs for buf[cur] complete
    __syncthreads();        // all waves' tiles visible; compute(it-1) done everywhere
    if (it + 1 < niter) issue(1 - cur, (it + 1) * BK);
    COMPUTE_STEP(cur)
  }
  epilogue<ACT>(acc, blk_m, blk_n, wm, wn, l15, lhi, bias, nullptr, out_bf16, gate, M, Nn);
}

// ---------------- weight conversion: f32 [K][N] -> bf16 K-major [N][K] ----------------
__global__ void cvt_transpose_bf16(const float* __restrict__ s, unsigned short* __restrict__ d,
                                   int K, int N) {
  int i = blockIdx.x * blockDim.x + threadIdx.x;
  if (i < K * N) {
    int k = i / N, n = i % N;
    d[(size_t)n * K + k] = f2bf(s[i]);
  }
}

// ---------------- A = g @ Wf + bf (Nn = 2) ----------------
__global__ __launch_bounds__(256)
void final_kernel(const unsigned short* __restrict__ g, const float* __restrict__ Wf,
                  const float* __restrict__ bf_, float* __restrict__ outA, int M) {
  __shared__ float w[D_H * 2];
  for (int i = threadIdx.x; i < D_H * 2; i += 256) w[i] = Wf[i];
  __syncthreads();
  int m = blockIdx.x * 256 + threadIdx.x;
  if (m >= M) return;
  float a0 = bf_[0], a1 = bf_[1];
  const unsigned short* gr = g + (size_t)m * D_H;
#pragma unroll 8
  for (int k = 0; k < D_H; ++k) {
    float gk = bf2f(gr[k]);
    a0 += gk * w[k * 2];
    a1 += gk * w[k * 2 + 1];
  }
  outA[(size_t)m * 2 + 0] = a0;
  outA[(size_t)m * 2 + 1] = a1;
}

extern "C" void kernel_launch(void* const* d_in, const int* in_sizes, int n_in,
                              void* d_out, int out_size, void* d_ws, size_t ws_size,
                              hipStream_t stream) {
  const float* X   = (const float*)d_in[0];
  const float* Wc  = (const float*)d_in[1];
  const float* bc  = (const float*)d_in[2];
  const float* Wv1 = (const float*)d_in[3];
  const float* bv1 = (const float*)d_in[4];
  const float* Wv2 = (const float*)d_in[5];
  const float* bv2 = (const float*)d_in[6];
  const float* Wu1 = (const float*)d_in[7];
  const float* bu1 = (const float*)d_in[8];
  const float* Wu2 = (const float*)d_in[9];
  const float* bu2 = (const float*)d_in[10];
  const float* Wf  = (const float*)d_in[11];
  const float* bf_ = (const float*)d_in[12];

  float* out_h = (float*)d_out;                       // [50000, 512]
  float* out_A = out_h + (size_t)N_ROWS * D_C;        // [50000, 2]

  // workspace layout (ushort = bf16); weights stored K-major [N][K]
  unsigned short* w    = (unsigned short*)d_ws;
  unsigned short* wWc  = w;                                   // 512 x 1024
  unsigned short* wWv1 = wWc  + (size_t)D_IN * D_C;           // 256 x 512
  unsigned short* wWv2 = wWv1 + (size_t)D_C  * D_H;           // 256 x 256
  unsigned short* wWu1 = wWv2 + (size_t)D_H  * D_H;           // 256 x 512
  unsigned short* wWu2 = wWu1 + (size_t)D_C  * D_H;           // 256 x 256
  unsigned short* hbf  = wWu2 + (size_t)D_H  * D_H;           // 50000 x 512
  unsigned short* tA   = hbf  + (size_t)N_ROWS * D_C;         // 50000 x 256
  unsigned short* tB   = tA   + (size_t)N_ROWS * D_H;         // 50000 x 256
  unsigned short* tC   = tB   + (size_t)N_ROWS * D_H;         // 50000 x 256

  {
    int n1 = D_IN * D_C, n2 = D_C * D_H, n3 = D_H * D_H;
    cvt_transpose_bf16<<<(n1 + 255) / 256, 256, 0, stream>>>(Wc,  wWc,  D_IN, D_C);
    cvt_transpose_bf16<<<(n2 + 255) / 256, 256, 0, stream>>>(Wv1, wWv1, D_C,  D_H);
    cvt_transpose_bf16<<<(n3 + 255) / 256, 256, 0, stream>>>(Wv2, wWv2, D_H,  D_H);
    cvt_transpose_bf16<<<(n2 + 255) / 256, 256, 0, stream>>>(Wu1, wWu1, D_C,  D_H);
    cvt_transpose_bf16<<<(n3 + 255) / 256, 256, 0, stream>>>(Wu2, wWu2, D_H,  D_H);
  }

  dim3 blk(THREADS);
  int gm = (N_ROWS + BM - 1) / BM;   // 391 row tiles

  // Grid: x = N tiles (fast-varying -> same-row blocks adjacent, A rows stay L2-hot),
  //       y = M tiles.
  // L1: h = relu(X @ Wc + bc)
  gemm_l1_kernel<<<dim3(D_C / BN, gm), blk, 0, stream>>>(
      X, wWc, bc, out_h, hbf, N_ROWS, D_C, D_IN);
  // L2: t1 = h @ Wv1 + bv1
  gemm_bf16_kernel<1><<<dim3(D_H / BN, gm), blk, 0, stream>>>(
      hbf, wWv1, bv1, tA, nullptr, N_ROWS, D_H, D_C);
  // L3: v = tanh(t1 @ Wv2 + bv2)
  gemm_bf16_kernel<2><<<dim3(D_H / BN, gm), blk, 0, stream>>>(
      tA, wWv2, bv2, tB, nullptr, N_ROWS, D_H, D_H);
  // L4: t2 = h @ Wu1 + bu1
  gemm_bf16_kernel<1><<<dim3(D_H / BN, gm), blk, 0, stream>>>(
      hbf, wWu1, bu1, tA, nullptr, N_ROWS, D_H, D_C);
  // L5: g = sigmoid(t2 @ Wu2 + bu2) * v
  gemm_bf16_kernel<3><<<dim3(D_H / BN, gm), blk, 0, stream>>>(
      tA, wWu2, bu2, tC, tB, N_ROWS, D_H, D_H);
  // L6: A = g @ Wf + bf
  final_kernel<<<(N_ROWS + 255) / 256, 256, 0, stream>>>(tC, Wf, bf_, out_A, N_ROWS);
}
